// VQ2Linear_32727650796016
// MI455X (gfx1250) — matrix operand order repaired and checked by reading, not compiled, not run on hardware
//
#include <hip/hip_runtime.h>
#include <hip/hip_bf16.h>

typedef __attribute__((ext_vector_type(16))) __bf16 v16bf;
typedef __attribute__((ext_vector_type(8)))  __bf16 v8bf;
typedef __attribute__((ext_vector_type(4)))  __bf16 v4bf;
typedef __attribute__((ext_vector_type(8)))  float  v8f;

#define B_TOKENS   131072
#define N_E        1024
#define E_DIM      64
#define CHUNK      256          // codebook rows staged in LDS per pass
#define NCHUNK     (N_E / CHUNK)
#define ESTRIDE    72           // 64 + 8 pad bf16 -> 144B row stride: 16B aligned, conflict-free
#define ROWS_PER_BLOCK 128      // 8 waves x 16 rows
#define WAVES      8

#define LOSS_OFF ((size_t)B_TOKENS * E_DIM)       // 8388608
#define IDX_OFF  (LOSS_OFF + 1)
#define VAR_OFF  (IDX_OFF + B_TOKENS)

// ---------------- zero the integer accumulators ----------------
__global__ void vq_zero(unsigned long long* ws_idx) {
    ws_idx[0] = 0ull;   // sum(idx)
    ws_idx[1] = 0ull;   // sum(idx^2)
}

// ---------------- main: WMMA distance + argmin + gather ----------------
__global__ __launch_bounds__(256) void vq_main(const float* __restrict__ z,
                                               const float* __restrict__ emb,
                                               float* __restrict__ out,
                                               double* __restrict__ ws_part,
                                               unsigned long long* __restrict__ ws_idx)
{
    __shared__ alignas(16) __bf16 s_emb[CHUNK][ESTRIDE]; // bf16(-2*emb)
    __shared__ float s_e2[CHUNK];                        // |e_n|^2 (fp32)
    __shared__ float s_redv[WAVES][16][16];
    __shared__ int   s_redi[WAVES][16][16];
    __shared__ int   s_bestn[WAVES][16];
    __shared__ float s_lsum;
    __shared__ unsigned long long s_sidx, s_sidx2;

    const int tid  = threadIdx.x;
    const int lane = tid & 31;
    const int wid  = tid >> 5;
    const int hh   = lane >> 4;    // half-wave selector (K-group / row-group)
    const int cls  = lane & 15;    // row (A) / column (B,D) class
    const int row0 = blockIdx.x * ROWS_PER_BLOCK + wid * 16;

    if (tid == 0) { s_lsum = 0.f; s_sidx = 0ull; s_sidx2 = 0ull; }

    // ---- A fragments: this wave's 16 z-rows, K=0..63, bf16, loaded once ----
    const float* zrow = z + (size_t)(row0 + cls) * E_DIM;
    float4 q0 = *(const float4*)(zrow + 8*hh + 0);
    float4 q1 = *(const float4*)(zrow + 8*hh + 4);
    float4 q2 = *(const float4*)(zrow + 8*hh + 16);
    float4 q3 = *(const float4*)(zrow + 8*hh + 20);
    float4 q4 = *(const float4*)(zrow + 8*hh + 32);
    float4 q5 = *(const float4*)(zrow + 8*hh + 36);
    float4 q6 = *(const float4*)(zrow + 8*hh + 48);
    float4 q7 = *(const float4*)(zrow + 8*hh + 52);
    v16bf a0, a1;
#define PK8(av, base, uu, ww) \
    av[(base)+0]=(__bf16)uu.x; av[(base)+1]=(__bf16)uu.y; av[(base)+2]=(__bf16)uu.z; av[(base)+3]=(__bf16)uu.w; \
    av[(base)+4]=(__bf16)ww.x; av[(base)+5]=(__bf16)ww.y; av[(base)+6]=(__bf16)ww.z; av[(base)+7]=(__bf16)ww.w;
    PK8(a0, 0, q0, q1) PK8(a0, 8, q2, q3)
    PK8(a1, 0, q4, q5) PK8(a1, 8, q6, q7)
#undef PK8

    float minv[8];
    int   mini[8];
#pragma unroll
    for (int r = 0; r < 8; ++r) { minv[r] = 3.0e38f; mini[r] = 0; }

    for (int ch = 0; ch < NCHUNK; ++ch) {
        __syncthreads();   // LDS safe to overwrite (also covers the tid==0 init)
        {   // stage: thread t converts codebook row (ch*CHUNK + t) -> bf16(-2*e), e2
            const float* er = emb + (size_t)(ch * CHUNK + tid) * E_DIM;
            float e2 = 0.f;
#pragma unroll
            for (int k = 0; k < E_DIM; k += 4) {
                float4 f = *(const float4*)(er + k);
                e2 += f.x*f.x + f.y*f.y + f.z*f.z + f.w*f.w;
                v4bf t;
                t[0] = (__bf16)(-2.f*f.x); t[1] = (__bf16)(-2.f*f.y);
                t[2] = (__bf16)(-2.f*f.z); t[3] = (__bf16)(-2.f*f.w);
                *(v4bf*)&s_emb[tid][k] = t;
            }
            s_e2[tid] = e2;
        }
        __syncthreads();

        for (int t = 0; t < CHUNK / 16; ++t) {
            const int ncol = t * 16 + cls;          // codebook row in chunk = my D-column
            const __bf16* bp = &s_emb[ncol][8 * hh];
            v8bf p0 = *(const v8bf*)(bp + 0);       // K chunk 0 (0..31)
            v8bf p1 = *(const v8bf*)(bp + 16);
            v8bf p2 = *(const v8bf*)(bp + 32);      // K chunk 1 (32..63)
            v8bf p3 = *(const v8bf*)(bp + 48);
            v16bf b0 = __builtin_shufflevector(p0, p1, 0,1,2,3,4,5,6,7,8,9,10,11,12,13,14,15);
            v16bf b1 = __builtin_shufflevector(p2, p3, 0,1,2,3,4,5,6,7,8,9,10,11,12,13,14,15);
            const float e2n = s_e2[ncol];
            v8f acc;
#pragma unroll
            for (int i = 0; i < 8; ++i) acc[i] = e2n;   // C = |e_n|^2
            // score = |e|^2 + z . (-2e)   (row-constant |z|^2 dropped)
            acc = __builtin_amdgcn_wmma_f32_16x16x32_bf16(false, a0, false, b0,
                                                          (short)0, acc, false, false);
            acc = __builtin_amdgcn_wmma_f32_16x16x32_bf16(false, a1, false, b1,
                                                          (short)0, acc, false, false);
            const int ng = ch * CHUNK + ncol;
#pragma unroll
            for (int r = 0; r < 8; ++r) {
                float v = acc[r];
                if (v < minv[r]) { minv[r] = v; mini[r] = ng; }
            }
        }
    }

    // ---- per-row argmin across the 16 column-lanes ----
#pragma unroll
    for (int r = 0; r < 8; ++r) {
        s_redv[wid][r + 8*hh][cls] = minv[r];
        s_redi[wid][r + 8*hh][cls] = mini[r];
    }
    __syncthreads();
    if (lane < 16) {
        float bv = s_redv[wid][lane][0];
        int   bn = s_redi[wid][lane][0];
        for (int j = 1; j < 16; ++j) {
            float v = s_redv[wid][lane][j];
            int   n = s_redi[wid][lane][j];
            if (v < bv || (v == bv && n < bn)) { bv = v; bn = n; }
        }
        s_bestn[wid][lane] = bn;
        out[IDX_OFF + row0 + lane] = (float)bn;       // idx emitted as float
        atomicAdd(&s_sidx,  (unsigned long long)bn);
        atomicAdd(&s_sidx2, (unsigned long long)bn * (unsigned)bn);
    }
    __syncthreads();

    // ---- gather z_q (exact fp32 from global) + loss partial ----
    float lsum = 0.f;
    for (int m = 0; m < 16; ++m) {
        const int n = s_bestn[wid][m];
        const size_t gr = (size_t)(row0 + m);
        float2 e  = *(const float2*)(emb + (size_t)n * E_DIM + 2*lane);
        float2 zz = *(const float2*)(z + gr * E_DIM + 2*lane);
        *(float2*)(out + gr * E_DIM + 2*lane) = e;    // z_q_st forward value == z_q
        float d0 = e.x - zz.x, d1 = e.y - zz.y;
        lsum += d0*d0 + d1*d1;
    }
    atomicAdd(&s_lsum, lsum);
    __syncthreads();
    if (tid == 0) {
        ws_part[blockIdx.x] = (double)s_lsum;         // deterministic partial
        atomicAdd(&ws_idx[0], s_sidx);                // exact integer sums
        atomicAdd(&ws_idx[1], s_sidx2);
    }
}

// ---------------- finalize: loss mean + unbiased variance ----------------
__global__ void vq_final(const double* __restrict__ part,
                         const unsigned long long* __restrict__ ws_idx,
                         float* __restrict__ out)
{
    double s = 0.0;
    for (int i = 0; i < B_TOKENS / ROWS_PER_BLOCK; ++i) s += part[i];
    const double loss = (1.0 + 0.25) * s / ((double)B_TOKENS * (double)E_DIM);
    const double sum  = (double)ws_idx[0];
    const double sum2 = (double)ws_idx[1];
    const double mean = sum / (double)B_TOKENS;
    const double var  = (sum2 - (double)B_TOKENS * mean * mean)
                        / ((double)B_TOKENS - 1.0);
    out[LOSS_OFF] = (float)loss;
    out[VAR_OFF]  = (float)var;
}

extern "C" void kernel_launch(void* const* d_in, const int* in_sizes, int n_in,
                              void* d_out, int out_size, void* d_ws, size_t ws_size,
                              hipStream_t stream) {
    const float* z   = (const float*)d_in[0];   // [131072, 64] fp32
    const float* emb = (const float*)d_in[1];   // [1024, 64]  fp32
    float* out = (float*)d_out;
    double* ws_part = (double*)d_ws;                                   // 1024 doubles
    unsigned long long* ws_idx = (unsigned long long*)((char*)d_ws + 8192); // 2 u64

    vq_zero<<<1, 1, 0, stream>>>(ws_idx);
    vq_main<<<dim3(B_TOKENS / ROWS_PER_BLOCK), dim3(256), 0, stream>>>(z, emb, out,
                                                                       ws_part, ws_idx);
    vq_final<<<1, 1, 0, stream>>>(ws_part, ws_idx, out);
}
